// GraphAttentionLayer_59648505806949
// MI455X (gfx1250) — matrix operand order repaired and checked by reading, not compiled, not run on hardware
//
#include <hip/hip_runtime.h>
#include <hip/hip_bf16.h>
#include <cstdint>

#define N_NODES 50000
#define E_EDGES 800000
#define IN_FEAT 256
#define EDGE_FEAT 64
#define OUT_FEAT 32
#define NHEAD 8
#define HF 256            // NHEAD * OUT_FEAT
#define ALPHA_SLOPE 0.2f
#define ROWS_PER_BLOCK 64

typedef __attribute__((ext_vector_type(16))) __bf16 v16bf;
typedef __attribute__((ext_vector_type(8)))  __bf16 v8bf;
typedef __attribute__((ext_vector_type(2)))  __bf16 v2bf;
typedef __attribute__((ext_vector_type(8)))  float  v8f;

// ---------------- helpers ----------------
__device__ __forceinline__ unsigned fkey(float f) {
    unsigned u = __float_as_uint(f);
    return (u & 0x80000000u) ? ~u : (u | 0x80000000u);
}
__device__ __forceinline__ float funkey(unsigned v) {
    unsigned u = (v & 0x80000000u) ? (v ^ 0x80000000u) : ~v;
    return __uint_as_float(u);
}

__global__ void fill_f32_kernel(float* p, float v, int n) {
    int i = blockIdx.x * blockDim.x + threadIdx.x;
    if (i < n) p[i] = v;
}
__global__ void fill_u32_kernel(unsigned* p, unsigned v, int n) {
    int i = blockIdx.x * blockDim.x + threadIdx.x;
    if (i < n) p[i] = v;
}

// ---------------- W -> bf16, transposed: Wt[col][k] ----------------
__global__ void convert_wt_kernel(const float* __restrict__ W, __bf16* __restrict__ Wt) {
    int i = blockIdx.x * blockDim.x + threadIdx.x;   // col*256 + k
    if (i >= IN_FEAT * HF) return;
    int col = i >> 8, k = i & 255;
    Wt[i] = (__bf16)W[(size_t)k * HF + col];
}

// ---------------- C[k][h] = sum_f W_e[k, h*32+f] * a_edge[h,f] ----------------
__global__ void cedge_kernel(const float* __restrict__ We_w, const float* __restrict__ a_edge,
                             float* __restrict__ C) {
    int i = blockIdx.x * blockDim.x + threadIdx.x;   // 0 .. 64*8-1
    if (i >= EDGE_FEAT * NHEAD) return;
    int k = i >> 3, hh = i & 7;
    float s = 0.f;
    #pragma unroll 8
    for (int f = 0; f < OUT_FEAT; ++f)
        s += We_w[k * HF + hh * OUT_FEAT + f] * a_edge[hh * OUT_FEAT + f];
    C[k * NHEAD + hh] = s;
}

// ---------------- Wh = h @ W  (bf16 WMMA, f32 accumulate) ----------------
// 64 rows x 256 cols per workgroup; 8 waves = rowtile(4) x colhalf(2), each wave
// owns 8 16x16 accumulators -> 8 WMMAs per K-step.
// Double-buffered LDS; B tile staged via GLOBAL_LOAD_ASYNC_TO_LDS_B128 from
// pre-transposed bf16 W so fragments are contiguous (ds_load_b128, not u16).
__global__ __launch_bounds__(256)
void gemm_wh_kernel(const float* __restrict__ h, const __bf16* __restrict__ Wt,
                    float* __restrict__ Wh) {
    __shared__ __align__(16) __bf16 As[2][ROWS_PER_BLOCK * 32];  // [row][k]  2x4KB
    __shared__ __align__(16) __bf16 Bs[2][HF * 32];              // [col][k]  2x16KB

    const int tid  = threadIdx.x;
    const int wave = tid >> 5;
    const int lane = tid & 31;
    const int row0 = blockIdx.x * ROWS_PER_BLOCK;
    const int rt   = wave >> 1;            // row tile 0..3 (16 rows each)
    const int ch   = (wave & 1) * 128;     // column half base

    v8f acc[8];
    #pragma unroll
    for (int t = 0; t < 8; ++t) acc[t] = (v8f){};

    auto stage = [&](int s, int buf) {
        const int k0 = s * 32;
        // async-copy B tile: Wt[col][k0..k0+32) -> Bs[buf][col][0..32), 16B chunks
        {
            const char* gbase = (const char*)Wt + (size_t)k0 * 2;
            unsigned    lb    = (unsigned)(size_t)(&Bs[buf][0]);
            #pragma unroll
            for (int t = 0; t < 4; ++t) {
                int idx = tid + t * 256;                 // chunk 0..1023
                int col = idx >> 2, q = idx & 3;
                unsigned long long gaddr =
                    (unsigned long long)(size_t)(gbase + (size_t)col * 512 + q * 16);
                unsigned laddr = lb + (unsigned)(idx * 16);
                asm volatile("global_load_async_to_lds_b128 %0, %1, off"
                             :: "v"(laddr), "v"(gaddr) : "memory");
            }
        }
        // stage A tile (f32 -> bf16), one float2 per thread x4
        #pragma unroll
        for (int t = 0; t < 4; ++t) {
            int idx = tid + t * 256;            // float2 slot 0..1023
            int r = idx >> 4, kp = (idx & 15) * 2;
            int gr = row0 + r;
            float2 v = (gr < N_NODES)
                     ? *(const float2*)(h + (size_t)gr * IN_FEAT + k0 + kp)
                     : make_float2(0.f, 0.f);
            v2bf pk; pk.x = (__bf16)v.x; pk.y = (__bf16)v.y;
            *(v2bf*)(&As[buf][r * 32 + kp]) = pk;
        }
    };

    stage(0, 0);
    asm volatile("s_wait_asynccnt 0x0" ::: "memory");
    __syncthreads();

    for (int s = 0; s < 8; ++s) {
        const int buf = s & 1;
        if (s < 7) stage(s + 1, buf ^ 1);   // overlap next-tile copy with compute

        // A fragment: lanes 0-15 -> K {0..15}; lanes 16-31 -> K {16..31}
        const int m   = rt * 16 + (lane & 15);
        const int kba = (lane < 16) ? 0 : 16;
        v8bf alo = *(const v8bf*)(&As[buf][m * 32 + kba]);
        v8bf ahi = *(const v8bf*)(&As[buf][m * 32 + 8 + kba]);
        v16bf af = __builtin_shufflevector(alo, ahi, 0, 1, 2, 3, 4, 5, 6, 7,
                                           8, 9, 10, 11, 12, 13, 14, 15);
        // B fragments: contiguous 32B per lane in [col][k] layout
        const int n   = lane & 15;
        const int kbb = (lane < 16) ? 0 : 16;
        #pragma unroll
        for (int t = 0; t < 8; ++t) {
            v16bf bfv = *(const v16bf*)(&Bs[buf][(ch + t * 16 + n) * 32 + kbb]);
            acc[t] = __builtin_amdgcn_wmma_f32_16x16x32_bf16(
                         false, af, false, bfv, (short)0, acc[t], false, false);
        }

        asm volatile("s_wait_asynccnt 0x0" ::: "memory");
        __syncthreads();
    }

    // store C tiles: vgpr r -> row (lane<16 ? r : 8+r), col = lane&15
    {
        const int n  = lane & 15;
        const int mb = (lane < 16) ? 0 : 8;
        #pragma unroll
        for (int t = 0; t < 8; ++t) {
            #pragma unroll
            for (int r = 0; r < 8; ++r) {
                int grow = row0 + rt * 16 + mb + r;
                if (grow < N_NODES)
                    Wh[(size_t)grow * HF + ch + t * 16 + n] = acc[t][r];
            }
        }
    }
}

// ---------------- s_src/s_dst[n,h] = <Wh[n,h,:], a_src/dst[h,:]> ----------------
__global__ void sdot_kernel(const float* __restrict__ Wh, const float* __restrict__ a_src,
                            const float* __restrict__ a_dst,
                            float* __restrict__ s_src, float* __restrict__ s_dst) {
    int i = blockIdx.x * blockDim.x + threadIdx.x;  // n*8 + h
    if (i >= N_NODES * NHEAD) return;
    int n = i >> 3, hh = i & 7;
    const float* row = Wh + (size_t)n * HF + hh * OUT_FEAT;
    const float* as  = a_src + hh * OUT_FEAT;
    const float* ad  = a_dst + hh * OUT_FEAT;
    float ss = 0.f, sd = 0.f;
    #pragma unroll 8
    for (int f = 0; f < OUT_FEAT; ++f) { float v = row[f]; ss += v * as[f]; sd += v * ad[f]; }
    s_src[i] = ss; s_dst[i] = sd;
}

// ---------------- s_edge = edge_features @ C  (warp per edge) ----------------
__global__ __launch_bounds__(256)
void sedge_kernel(const float* __restrict__ ef, const float* __restrict__ C,
                  float* __restrict__ score) {
    __shared__ float cs[EDGE_FEAT * NHEAD];
    for (int i = threadIdx.x; i < EDGE_FEAT * NHEAD; i += blockDim.x) cs[i] = C[i];
    __syncthreads();
    int e    = blockIdx.x * (blockDim.x >> 5) + (threadIdx.x >> 5);
    int lane = threadIdx.x & 31;
    if (e >= E_EDGES) return;
    float2 v = ((const float2*)(ef + (size_t)e * EDGE_FEAT))[lane];  // k = 2*lane, 2*lane+1
    float p[NHEAD];
    #pragma unroll
    for (int hh = 0; hh < NHEAD; ++hh)
        p[hh] = v.x * cs[(2 * lane) * NHEAD + hh] + v.y * cs[(2 * lane + 1) * NHEAD + hh];
    #pragma unroll
    for (int off = 16; off; off >>= 1) {
        #pragma unroll
        for (int hh = 0; hh < NHEAD; ++hh) p[hh] += __shfl_xor(p[hh], off, 32);
    }
    if (lane == 0) {
        #pragma unroll
        for (int hh = 0; hh < NHEAD; ++hh) score[(size_t)e * NHEAD + hh] = p[hh];
    }
}

// ---------------- score = leaky_relu(s_src[src]+s_dst[dst]+s_edge); segment max ----------------
__global__ void score_kernel(const float* __restrict__ s_src, const float* __restrict__ s_dst,
                             float* __restrict__ score, const int* __restrict__ src,
                             const int* __restrict__ dst, unsigned* __restrict__ mkey) {
    int i = blockIdx.x * blockDim.x + threadIdx.x;  // e*8 + h
    if (i >= E_EDGES * NHEAD) return;
    int e = i >> 3, hh = i & 7;
    int s = src[e], d = dst[e];
    float x = s_src[s * NHEAD + hh] + s_dst[d * NHEAD + hh] + score[i];
    x = (x > 0.f) ? x : ALPHA_SLOPE * x;
    score[i] = x;
    atomicMax(&mkey[d * NHEAD + hh], fkey(x));
}

// ---------------- e = exp(score - m[dst]); denom[dst] += e ----------------
__global__ void expsum_kernel(float* __restrict__ score, const int* __restrict__ dst,
                              const unsigned* __restrict__ mkey, float* __restrict__ denom) {
    int i = blockIdx.x * blockDim.x + threadIdx.x;
    if (i >= E_EDGES * NHEAD) return;
    int e = i >> 3, hh = i & 7;
    int d = dst[e];
    unsigned k = mkey[d * NHEAD + hh];
    float m = (k == 0u) ? 0.f : funkey(k);        // empty-segment fixup (reference: isfinite->0)
    float ex = __expf(score[i] - m);
    score[i] = ex;
    atomicAdd(&denom[d * NHEAD + hh], ex);
}

// ---------------- h_new[dst] += Wh[src] * (e/denom[dst])  (warp per edge) ----------------
__global__ __launch_bounds__(256)
void scatter_kernel(const float* __restrict__ Wh, const float* __restrict__ escore,
                    const float* __restrict__ denom, const int* __restrict__ src,
                    const int* __restrict__ dst, float* __restrict__ out) {
    int e    = blockIdx.x * (blockDim.x >> 5) + (threadIdx.x >> 5);
    int lane = threadIdx.x & 31;
    if (e >= E_EDGES) return;
    int s = src[e], d = dst[e];
    const float* wrow = Wh + (size_t)s * HF;
    float*       orow = out + (size_t)d * HF;
    #pragma unroll
    for (int hh = 0; hh < NHEAD; ++hh) {
        float a = escore[(size_t)e * NHEAD + hh] / denom[d * NHEAD + hh];
        atomicAdd(&orow[hh * OUT_FEAT + lane], wrow[hh * OUT_FEAT + lane] * a);
    }
}

// ---------------- out = elu(out) ----------------
__global__ void elu_kernel(float* __restrict__ out, int n) {
    int i = blockIdx.x * blockDim.x + threadIdx.x;
    if (i < n) { float x = out[i]; out[i] = (x > 0.f) ? x : (__expf(x) - 1.f); }
}

// =====================================================================
extern "C" void kernel_launch(void* const* d_in, const int* in_sizes, int n_in,
                              void* d_out, int out_size, void* d_ws, size_t ws_size,
                              hipStream_t stream) {
    const float* h    = (const float*)d_in[0];   // N x 256
    const float* ef   = (const float*)d_in[1];   // E x 64
    const float* W    = (const float*)d_in[2];   // 256 x 256
    const float* W_e  = (const float*)d_in[3];   // 64 x 256
    const float* a_s  = (const float*)d_in[4];   // 8 x 32
    const float* a_d  = (const float*)d_in[5];   // 8 x 32
    const float* a_e  = (const float*)d_in[6];   // 8 x 32
    const int*   src  = (const int*)d_in[7];     // E
    const int*   dst  = (const int*)d_in[8];     // E
    float* out = (float*)d_out;                  // N x 256

    // workspace layout (256B aligned slabs)
    char* ws = (char*)d_ws;
    size_t off = 0;
    auto alloc = [&](size_t bytes) { char* p = ws + off; off = (off + bytes + 255) & ~(size_t)255; return p; };
    float*    Wh     = (float*)alloc((size_t)N_NODES * HF * 4);      // 51.2 MB
    float*    score  = (float*)alloc((size_t)E_EDGES * NHEAD * 4);   // 25.6 MB
    float*    s_src  = (float*)alloc((size_t)N_NODES * NHEAD * 4);
    float*    s_dst  = (float*)alloc((size_t)N_NODES * NHEAD * 4);
    unsigned* mkey   = (unsigned*)alloc((size_t)N_NODES * NHEAD * 4);
    float*    denom  = (float*)alloc((size_t)N_NODES * NHEAD * 4);
    float*    Cmat   = (float*)alloc((size_t)EDGE_FEAT * NHEAD * 4);
    __bf16*   Wt     = (__bf16*)alloc((size_t)IN_FEAT * HF * 2);     // transposed bf16 W

    const int TPB = 256;
    // init accumulators (d_out / d_ws are poisoned, not re-zeroed between replays)
    fill_f32_kernel<<<(N_NODES * HF + TPB - 1) / TPB, TPB, 0, stream>>>(out, 0.f, N_NODES * HF);
    fill_u32_kernel<<<(N_NODES * NHEAD + TPB - 1) / TPB, TPB, 0, stream>>>(mkey, 0u, N_NODES * NHEAD);
    fill_f32_kernel<<<(N_NODES * NHEAD + TPB - 1) / TPB, TPB, 0, stream>>>(denom, 0.f, N_NODES * NHEAD);

    // tiny precomputes
    convert_wt_kernel<<<(IN_FEAT * HF + TPB - 1) / TPB, TPB, 0, stream>>>(W, Wt);
    cedge_kernel<<<(EDGE_FEAT * NHEAD + TPB - 1) / TPB, TPB, 0, stream>>>(W_e, a_e, Cmat);

    // node projection via WMMA (64-row blocks, double-buffered async B staging)
    gemm_wh_kernel<<<(N_NODES + ROWS_PER_BLOCK - 1) / ROWS_PER_BLOCK, 256, 0, stream>>>(h, Wt, Wh);

    // per-node attention dot products
    sdot_kernel<<<(N_NODES * NHEAD + TPB - 1) / TPB, TPB, 0, stream>>>(Wh, a_s, a_d, s_src, s_dst);

    // fused edge projection -> s_edge (stored into `score`)
    sedge_kernel<<<(E_EDGES + 7) / 8, 256, 0, stream>>>(ef, Cmat, score);

    // leaky-relu score + segment max
    score_kernel<<<(E_EDGES * NHEAD + TPB - 1) / TPB, TPB, 0, stream>>>(s_src, s_dst, score, src, dst, mkey);

    // exp + segment sum
    expsum_kernel<<<(E_EDGES * NHEAD + TPB - 1) / TPB, TPB, 0, stream>>>(score, dst, mkey, denom);

    // weighted message scatter-add
    scatter_kernel<<<(E_EDGES + 7) / 8, 256, 0, stream>>>(Wh, score, denom, src, dst, out);

    // final ELU
    elu_kernel<<<(N_NODES * HF + TPB - 1) / TPB, TPB, 0, stream>>>(out, N_NODES * HF);
}
